// SelfAttention_19138374271186
// MI455X (gfx1250) — matrix operand order repaired and checked by reading, not compiled, not run on hardware
//
#include <hip/hip_runtime.h>

// ---------------------------------------------------------------------------
// MI455X (gfx1250) self-attention: flash-attention with bf16 WMMA + fc_out GEMM
//
// Stage 1 (fa_kernel): per (n, head, 128-row q-tile) workgroup, 8 waves.
//   Each wave keeps a 16x64 Q tile as bf16 A-fragments in VGPRs, streams
//   64-row K/V tiles through LDS (f32->bf16 at store), stages the int32 mask
//   tile with global_load_async_to_lds_b128 (ASYNCcnt) when available,
//   runs online softmax, accumulates O in f32.
// Stage 2 (gemm_kernel): ctx[8192x1024] @ w_out^T + b_out, 128x128 tiles.
// ---------------------------------------------------------------------------

typedef __attribute__((ext_vector_type(16))) __bf16 bf16x16;
typedef __attribute__((ext_vector_type(8)))  float  f32x8;

#define S_LEN   2048
#define EMB     1024
#define NHEADS  16
#define HDIM    64
#define BQ      128      // query rows per workgroup (8 waves x 16)
#define BK      64       // key/value rows per tile
#define KPAD    72       // padded LDS row stride (elements) to spread banks
#define MPAD    68       // padded mask row stride (int32 elements)
#define SCALE   0.03125f // 1/sqrt(1024)

#if defined(__HIP_DEVICE_COMPILE__) && defined(__has_builtin)
#  if __has_builtin(__builtin_amdgcn_global_load_async_to_lds_b128)
#    define HAVE_ASYNC_LDS 1
#  endif
#endif
#ifndef HAVE_ASYNC_LDS
#  define HAVE_ASYNC_LDS 0
#endif

#if HAVE_ASYNC_LDS
// Builtin signature (from compiler diagnostic): parameters are pointers to
// 16-byte int vectors in AS1 (global) / AS3 (LDS).
typedef int v4i __attribute__((vector_size(16)));
typedef __attribute__((address_space(1))) v4i g_v4i;
typedef __attribute__((address_space(3))) v4i l_v4i;
#define ASYNC_CP_B128(gsrc, ldst)                                     \
    __builtin_amdgcn_global_load_async_to_lds_b128(                   \
        (g_v4i*)(gsrc), (l_v4i*)(ldst), 0, 0)
__device__ __forceinline__ void wait_async0() {
#if __has_builtin(__builtin_amdgcn_s_wait_asynccnt)
    __builtin_amdgcn_s_wait_asynccnt(0);
#else
    asm volatile("s_wait_asynccnt 0" ::: "memory");
#endif
}
#endif

__device__ __forceinline__ __bf16 to_bf16(float f) { return (__bf16)f; }

// ===========================================================================
// Kernel 1: flash attention  ->  ctx[n][s][h*64+d]  (f32, in workspace)
// ===========================================================================
__global__ __launch_bounds__(256) void fa_kernel(
    const float* __restrict__ Vg, const float* __restrict__ Kg,
    const float* __restrict__ Qg, const int*  __restrict__ Mg,
    float* __restrict__ ctx)
{
    __shared__ __bf16 Ks[BK][KPAD];         // K tile, row-major [kseq][d]
    __shared__ __bf16 Vt[HDIM][KPAD];       // V tile, transposed [d][kseq]
    __shared__ __bf16 Ps[8][16][KPAD];      // per-wave P scratch [row][kseq]
#if HAVE_ASYNC_LDS
    __shared__ int    Msk[BQ][MPAD];        // mask tile [qrow_local][kcol]
#endif

    const int tid  = threadIdx.x;
    const int wave = tid >> 5;
    const int lane = tid & 31;
    const int ln   = lane & 15;             // column / row-within-16
    const int lh   = lane >> 4;             // half-wave select

    const int blk  = blockIdx.x;            // N*H*(S/BQ) = 1024 blocks
    const int qt   = blk & 15;
    const int head = (blk >> 4) & 15;
    const int n    = blk >> 8;

    const int qbase = qt * BQ + wave * 16;  // this wave's 16 query rows

    const float* Qh = Qg + (size_t)n * S_LEN * EMB + head * HDIM;
    const float* Kh = Kg + (size_t)n * S_LEN * EMB + head * HDIM;
    const float* Vh = Vg + (size_t)n * S_LEN * EMB + head * HDIM;
    const int*   Mh = Mg + (size_t)n * S_LEN * S_LEN;

    // ---- persistent Q A-fragments (16 rows x 64 d -> two 16x32 bf16 tiles)
    bf16x16 qfrag[2];
    {
        const float* qrow = Qh + (size_t)(qbase + ln) * EMB;
        #pragma unroll
        for (int c = 0; c < 2; ++c)
            #pragma unroll
            for (int i = 0; i < 16; ++i) {
                int d = 32 * c + 16 * lh + i;
                qfrag[c][i] = to_bf16(qrow[d]);
            }
    }

    // ---- online softmax state (per lane: 8 rows, replicated over 16 cols)
    float rmax[8], rsum[8];
    f32x8 oacc[4];
    #pragma unroll
    for (int r = 0; r < 8; ++r) { rmax[r] = -1.0e30f; rsum[r] = 0.0f; }
    #pragma unroll
    for (int j = 0; j < 4; ++j)
        #pragma unroll
        for (int r = 0; r < 8; ++r) oacc[j][r] = 0.0f;

    for (int kt = 0; kt < S_LEN / BK; ++kt) {
#if HAVE_ASYNC_LDS
        // ---- async-stage 128x64 int32 mask tile (coalesced b128 per lane)
        {
            const int* msrc = Mh + (size_t)(qt * BQ) * S_LEN + kt * BK;
            #pragma unroll
            for (int it = 0; it < 8; ++it) {
                int flat = (it * 256 + tid) * 4;     // int index within tile
                int row  = flat >> 6;
                int col  = flat & 63;
                ASYNC_CP_B128(msrc + (size_t)row * S_LEN + col, &Msk[row][col]);
            }
        }
#endif
        // ---- cooperative tile load: K row-major, V transposed, f32 -> bf16
        {
            const int row = tid >> 2;            // 0..63
            const int dc  = (tid & 3) * 16;      // 0,16,32,48
            const float* krow = Kh + (size_t)(kt * BK + row) * EMB + dc;
            const float* vrow = Vh + (size_t)(kt * BK + row) * EMB + dc;
            #pragma unroll
            for (int i = 0; i < 16; ++i) Ks[row][dc + i] = to_bf16(krow[i]);
            #pragma unroll
            for (int i = 0; i < 16; ++i) Vt[dc + i][row] = to_bf16(vrow[i]);
            if (kt + 1 < S_LEN / BK) {          // prefetch next K/V tiles
                __builtin_prefetch(krow + (size_t)BK * EMB, 0, 1);
                __builtin_prefetch(vrow + (size_t)BK * EMB, 0, 1);
            }
        }
#if HAVE_ASYNC_LDS
        wait_async0();
#endif
        __syncthreads();

        // ---- S = Q * K^T   (16 x 64 scores, 4 col-blocks x 2 k-chunks)
        f32x8 sblk[4];
        #pragma unroll
        for (int b = 0; b < 4; ++b) {
            f32x8 acc;
            #pragma unroll
            for (int r = 0; r < 8; ++r) acc[r] = 0.0f;
            #pragma unroll
            for (int c = 0; c < 2; ++c) {
                bf16x16 bfrag;
                const __bf16* src = &Ks[b * 16 + ln][32 * c + 16 * lh];
                #pragma unroll
                for (int i = 0; i < 16; ++i) bfrag[i] = src[i];
                acc = __builtin_amdgcn_wmma_f32_16x16x32_bf16(
                        false, qfrag[c], false, bfrag, (short)0, acc,
                        false, false);
            }
            sblk[b] = acc;
        }

        // ---- mask (then scale, matching reference order of operations)
        #pragma unroll
        for (int b = 0; b < 4; ++b) {
#if HAVE_ASYNC_LDS
            #pragma unroll
            for (int r = 0; r < 8; ++r) {
                int mv = Msk[wave * 16 + 8 * lh + r][b * 16 + ln];
                float s = sblk[b][r] * SCALE;
                sblk[b][r] = mv ? s : -1.0e20f;
            }
#else
            const int col = kt * BK + b * 16 + ln;
            const int* mcol = Mh + (size_t)(qbase + 8 * lh) * S_LEN + col;
            #pragma unroll
            for (int r = 0; r < 8; ++r) {
                int mv = mcol[(size_t)r * S_LEN];
                float s = sblk[b][r] * SCALE;
                sblk[b][r] = mv ? s : -1.0e20f;
            }
#endif
        }

        // ---- online softmax update (row reductions across 16-lane groups)
        #pragma unroll
        for (int r = 0; r < 8; ++r) {
            float tmax = sblk[0][r];
            #pragma unroll
            for (int b = 1; b < 4; ++b) tmax = fmaxf(tmax, sblk[b][r]);
            #pragma unroll
            for (int off = 1; off < 16; off <<= 1)
                tmax = fmaxf(tmax, __shfl_xor(tmax, off, 32));
            float nmax = fmaxf(rmax[r], tmax);
            float corr = __expf(rmax[r] - nmax);
            rmax[r] = nmax;
            float lsum = 0.0f;
            #pragma unroll
            for (int b = 0; b < 4; ++b) {
                float p = __expf(sblk[b][r] - nmax);
                sblk[b][r] = p;
                lsum += p;
            }
            #pragma unroll
            for (int off = 1; off < 16; off <<= 1)
                lsum += __shfl_xor(lsum, off, 32);
            rsum[r] = rsum[r] * corr + lsum;
            #pragma unroll
            for (int j = 0; j < 4; ++j) oacc[j][r] *= corr;
        }

        // ---- re-layout P through per-wave LDS (C/D layout -> A layout)
        #pragma unroll
        for (int b = 0; b < 4; ++b)
            #pragma unroll
            for (int r = 0; r < 8; ++r)
                Ps[wave][r + 8 * lh][b * 16 + ln] = to_bf16(sblk[b][r]);

        bf16x16 pfrag[2];
        #pragma unroll
        for (int c = 0; c < 2; ++c)
            #pragma unroll
            for (int i = 0; i < 16; ++i)
                pfrag[c][i] =
                    Ps[wave][ln][32 * c + 16 * lh + i];

        // ---- O += P * V   (V read transposed so B k-dim is contiguous)
        #pragma unroll
        for (int j = 0; j < 4; ++j)
            #pragma unroll
            for (int c = 0; c < 2; ++c) {
                bf16x16 vfrag;
                const __bf16* src = &Vt[j * 16 + ln][32 * c + 16 * lh];
                #pragma unroll
                for (int i = 0; i < 16; ++i) vfrag[i] = src[i];
                oacc[j] = __builtin_amdgcn_wmma_f32_16x16x32_bf16(
                            false, pfrag[c], false, vfrag, (short)0, oacc[j],
                            false, false);
            }
        __syncthreads();   // guard K/V/mask tiles before next overwrite
    }

    // ---- finalize: divide by row sums, write ctx (f32)
    float* crow = ctx + ((size_t)n * S_LEN + qbase + 8 * lh) * EMB + head * HDIM;
    #pragma unroll
    for (int r = 0; r < 8; ++r) {
        float inv = 1.0f / fmaxf(rsum[r], 1.0e-30f);
        #pragma unroll
        for (int j = 0; j < 4; ++j)
            crow[(size_t)r * EMB + j * 16 + ln] = oacc[j][r] * inv;
    }
}

// ===========================================================================
// Kernel 2: out[i][j] = sum_k ctx[i][k] * w_out[j][k] + b_out[j]
//   128x128 output tiles, 32-deep k-steps through LDS, bf16 WMMA.
// ===========================================================================
#define KC    32
#define KCPAD 40

__global__ __launch_bounds__(256) void gemm_kernel(
    const float* __restrict__ ctx, const float* __restrict__ Wg,
    const float* __restrict__ Bg, float* __restrict__ Yg)
{
    __shared__ __bf16 As[128][KCPAD];   // ctx tile   [row][k]
    __shared__ __bf16 Bs[128][KCPAD];   // w_out tile [col][k] (W rows = cols)

    const int tid  = threadIdx.x;
    const int wave = tid >> 5;
    const int lane = tid & 31;
    const int ln   = lane & 15;
    const int lh   = lane >> 4;

    const int rowbase = blockIdx.x * 128;   // 64 blocks over M = 8192
    const int colbase = blockIdx.y * 128;   // 8 blocks over N = 1024

    f32x8 acc[8];
    #pragma unroll
    for (int j = 0; j < 8; ++j)
        #pragma unroll
        for (int r = 0; r < 8; ++r) acc[j][r] = 0.0f;

    for (int kt = 0; kt < EMB / KC; ++kt) {
        {
            const int row = tid >> 1;             // 0..127
            const int ks  = (tid & 1) * 16;       // 0 or 16
            const float* arow = ctx + (size_t)(rowbase + row) * EMB + kt * KC + ks;
            const float* brow = Wg  + (size_t)(colbase + row) * EMB + kt * KC + ks;
            #pragma unroll
            for (int i = 0; i < 16; ++i) As[row][ks + i] = to_bf16(arow[i]);
            #pragma unroll
            for (int i = 0; i < 16; ++i) Bs[row][ks + i] = to_bf16(brow[i]);
            if (kt + 1 < EMB / KC) {              // prefetch next k-step tiles
                __builtin_prefetch(arow + KC, 0, 1);
                __builtin_prefetch(brow + KC, 0, 1);
            }
        }
        __syncthreads();

        bf16x16 afrag;
        #pragma unroll
        for (int i = 0; i < 16; ++i)
            afrag[i] = As[16 * wave + ln][16 * lh + i];

        #pragma unroll
        for (int j = 0; j < 8; ++j) {
            bf16x16 bfrag;
            const __bf16* src = &Bs[j * 16 + ln][16 * lh];
            #pragma unroll
            for (int i = 0; i < 16; ++i) bfrag[i] = src[i];
            acc[j] = __builtin_amdgcn_wmma_f32_16x16x32_bf16(
                        false, afrag, false, bfrag, (short)0, acc[j],
                        false, false);
        }
        __syncthreads();
    }

    // ---- epilogue: bias + store
    #pragma unroll
    for (int j = 0; j < 8; ++j) {
        const int col = colbase + j * 16 + ln;
        const float bias = Bg[col];
        #pragma unroll
        for (int r = 0; r < 8; ++r) {
            const int row = rowbase + 16 * wave + 8 * lh + r;
            Yg[(size_t)row * EMB + col] = acc[j][r] + bias;
        }
    }
}

// ===========================================================================
extern "C" void kernel_launch(void* const* d_in, const int* in_sizes, int n_in,
                              void* d_out, int out_size, void* d_ws, size_t ws_size,
                              hipStream_t stream) {
    const float* v    = (const float*)d_in[0];
    const float* k    = (const float*)d_in[1];
    const float* q    = (const float*)d_in[2];
    const int*   mask = (const int*)  d_in[3];
    const float* w    = (const float*)d_in[4];
    const float* b    = (const float*)d_in[5];
    float* out = (float*)d_out;
    float* ctx = (float*)d_ws;   // N*S*E f32 = 32 MiB attention context

    // N*H*(S/BQ) = 4*16*16 = 1024 workgroups, 8 waves each
    fa_kernel<<<dim3(1024), dim3(256), 0, stream>>>(v, k, q, mask, ctx);
    // (8192/128) x (1024/128) output tiles
    gemm_kernel<<<dim3(64, 8), dim3(256), 0, stream>>>(ctx, w, b, out);
}
